// SelfAttnV2_63118839382025
// MI455X (gfx1250) — matrix operand order repaired and checked by reading, not compiled, not run on hardware
//
#include <hip/hip_runtime.h>
#include <hip/hip_bf16.h>

// ---------------------------------------------------------------------------
// Self-attention (B=4, S=2048, H=1024, single head, causal) for gfx1250.
//   1) cvt: fp32 -> bf16 for X, Wq, Wk, Wv
//   2) qkv: LDS-tiled 128x128 block GEMM, async-to-LDS double-buffered
//           staging, 64x32 register tile per wave (8 wmma / 6 frag loads)
//   3) attn: flash-style causal attention; Q tile staged via TDM
//           (tensor_load_to_lds + s_wait_tensorcnt), scores/PV on
//           v_wmma_f32_16x16x32_bf16
// Workspace (~70 MB): Xb 16M | Wqb/Wkb/Wvb 2M each | Qb 16M | Kb 16M | Vtb 16M
// ---------------------------------------------------------------------------

typedef __attribute__((ext_vector_type(16))) __bf16 bf16x16;
typedef __attribute__((ext_vector_type(8)))  __bf16 bf16x8;
typedef __attribute__((ext_vector_type(4)))  __bf16 bf16x4;
typedef __attribute__((ext_vector_type(8)))  float  f32x8;
typedef __attribute__((ext_vector_type(4)))  unsigned uint32x4;
typedef __attribute__((ext_vector_type(8)))  unsigned uint32x8;

#define S_LEN 2048
#define HD    1024
#define BATCH 4
#define KC    128
#define SCALE 0.03125f   /* 1/sqrt(1024) */

// ---- CDNA5 async global->LDS copy (ASYNCcnt path, ISA 15.18.3 op 98) ------
__device__ __forceinline__ void async_copy_b128(void* lds_ptr, const void* gptr) {
  unsigned lds_off = (unsigned)(unsigned long long)lds_ptr;
  unsigned long long ga = (unsigned long long)gptr;
  asm volatile("global_load_async_to_lds_b128 %0, %1, off"
               :: "v"(lds_off), "v"(ga) : "memory");
}
#define WAIT_ASYNC_0 asm volatile("s_wait_asynccnt 0" ::: "memory")
#define WAIT_ASYNC_4 asm volatile("s_wait_asynccnt 4" ::: "memory")

// ---- CDNA5 Tensor Data Mover: 2D tile load, D# per ISA ch.8 ---------------
// rows x cols elements of 2-byte data, row stride `stride` elements.
__device__ __forceinline__ void tdm_load_2d_b16(void* lds_ptr, const void* gptr,
                                                unsigned rows, unsigned cols,
                                                unsigned stride) {
  unsigned long long ga = (unsigned long long)gptr;
  unsigned lds = (unsigned)(unsigned long long)lds_ptr;
  uint32x4 g0;
  g0[0] = 1u;                                    // count=1 (valid), user mode
  g0[1] = lds;                                   // lds_addr
  g0[2] = (unsigned)ga;                          // global_addr[31:0]
  g0[3] = (unsigned)(ga >> 32) | (2u << 30);     // global_addr[56:32] | type=2
  uint32x8 g1;
  g1[0] = 1u << 16;                              // data_size=1 (2 bytes)
  g1[1] = cols << 16;                            // tensor_dim0[15:0]
  g1[2] = (cols >> 16) | (rows << 16);           // tensor_dim0[31:16]|dim1 lo
  g1[3] = cols << 16;                            // dim1 hi=0 | tile_dim0
  g1[4] = rows;                                  // tile_dim1 | tile_dim2=0
  g1[5] = stride;                                // tensor_dim0_stride[31:0]
  g1[6] = 0u;                                    // stride hi | dim1_stride lo
  g1[7] = 0u;
  asm volatile("tensor_load_to_lds %0, %1" :: "s"(g0), "s"(g1) : "memory");
}

// ---- fragment loaders (WMMA 16x16x32 bf16 VGPR layouts, ISA 7.12.2) -------
__device__ __forceinline__ bf16x16 load_frag_a(const __bf16* src, int ld) {
  int lane = threadIdx.x & 31;
  int m    = lane & 15;
  int koff = (lane >> 4) << 3;
  const __bf16* p = src + m * ld + koff;
  bf16x8 lo = *(const bf16x8*)(p);
  bf16x8 hi = *(const bf16x8*)(p + 16);
  bf16x16 a;
#pragma unroll
  for (int i = 0; i < 8; ++i) { a[i] = lo[i]; a[i + 8] = hi[i]; }
  return a;
}
__device__ __forceinline__ bf16x16 load_frag_b(const __bf16* src, int ld) {
  int lane = threadIdx.x & 31;
  int n    = lane & 15;
  int kb2  = (lane >> 4) << 3;
  const __bf16* p = src + n * ld + kb2;
  bf16x8 lo = *(const bf16x8*)(p);
  bf16x8 hi = *(const bf16x8*)(p + 16);
  bf16x16 b;
#pragma unroll
  for (int i = 0; i < 8; ++i) { b[i] = lo[i]; b[i + 8] = hi[i]; }
  return b;
}
__device__ __forceinline__ f32x8 wmma_bf16(bf16x16 a, bf16x16 b, f32x8 c) {
  return __builtin_amdgcn_wmma_f32_16x16x32_bf16(
      false, a, false, b, (short)0, c, false, false);
}

// ---------------------------------------------------------------------------
__global__ __launch_bounds__(256) void SA_cvt_kernel(const float* __restrict__ in,
                                                     __bf16* __restrict__ out,
                                                     int n4) {
  int i = blockIdx.x * 256 + threadIdx.x;
  if (i < n4) {
    float4 v = ((const float4*)in)[i];
    bf16x4 o;
    o[0] = (__bf16)v.x; o[1] = (__bf16)v.y; o[2] = (__bf16)v.z; o[3] = (__bf16)v.w;
    ((bf16x4*)out)[i] = o;
  }
}

// ---------------------------------------------------------------------------
// Fused QKV projection, LDS-tiled, async double-buffered.
__global__ __launch_bounds__(256) void SA_qkv_kernel(
    const __bf16* __restrict__ X,
    const __bf16* __restrict__ Wq, const __bf16* __restrict__ Wk,
    const __bf16* __restrict__ Wv,
    __bf16* __restrict__ Qo, __bf16* __restrict__ Ko,
    __bf16* __restrict__ Vt) {
  __shared__ __bf16 Xs[2][128 * 32];
  __shared__ __bf16 Ws[2][128 * 32];

  int tid  = threadIdx.x;
  int wave = tid >> 5;
  int lane = tid & 31;
  int bx   = blockIdx.x;
  int mat  = bx / 512;
  int rem  = bx - mat * 512;
  int mblk = (rem >> 3) << 7;
  int nblk = (rem & 7) << 7;
  const __bf16* W = (mat == 0) ? Wq : ((mat == 1) ? Wk : Wv);
  int wm = wave & 1;
  int wn = wave >> 1;

  const __bf16* Xbase = X + (long)mblk * HD;
  const __bf16* Wbase = W + (long)nblk * HD;

  f32x8 acc[4][2];
#pragma unroll
  for (int mt = 0; mt < 4; ++mt)
#pragma unroll
    for (int nt = 0; nt < 2; ++nt)
      acc[mt][nt] = (f32x8){0.f,0.f,0.f,0.f,0.f,0.f,0.f,0.f};

  auto stage = [&](int ks, int sel) {
#pragma unroll
    for (int i = tid; i < 512; i += 256) {
      int row = i >> 2, seg = (i & 3) << 3;
      async_copy_b128(&Xs[sel][row * 32 + seg],
                      Xbase + ks * 32 + (long)row * HD + seg);
      async_copy_b128(&Ws[sel][row * 32 + seg],
                      Wbase + ks * 32 + (long)row * HD + seg);
    }
  };

  stage(0, 0);
  for (int ks = 0; ks < 32; ++ks) {
    int sel = ks & 1;
    if (ks + 1 < 32) { stage(ks + 1, sel ^ 1); WAIT_ASYNC_4; }
    else             { WAIT_ASYNC_0; }
    __syncthreads();

    bf16x16 afr[4], bfr[2];
#pragma unroll
    for (int mt = 0; mt < 4; ++mt)
      afr[mt] = load_frag_a(&Xs[sel][(wm * 64 + mt * 16) * 32], 32);
#pragma unroll
    for (int nt = 0; nt < 2; ++nt)
      bfr[nt] = load_frag_b(&Ws[sel][(wn * 32 + nt * 16) * 32], 32);
#pragma unroll
    for (int mt = 0; mt < 4; ++mt)
#pragma unroll
      for (int nt = 0; nt < 2; ++nt)
        acc[mt][nt] = wmma_bf16(afr[mt], bfr[nt], acc[mt][nt]);
    __syncthreads();
  }

  int n    = lane & 15;
  int moff = (lane >> 4) << 3;
#pragma unroll
  for (int mt = 0; mt < 4; ++mt) {
#pragma unroll
    for (int nt = 0; nt < 2; ++nt) {
      int m0 = mblk + wm * 64 + mt * 16;
      int n0 = nblk + wn * 32 + nt * 16;
      if (mat == 2) {
#pragma unroll
        for (int r = 0; r < 8; ++r) {
          int mg = m0 + moff + r;
          int bb = mg >> 11;
          int s  = mg & (S_LEN - 1);
          Vt[(long)bb * HD * S_LEN + (long)(n0 + n) * S_LEN + s] =
              (__bf16)acc[mt][nt][r];
        }
      } else {
        __bf16* O = (mat == 0) ? Qo : Ko;
#pragma unroll
        for (int r = 0; r < 8; ++r)
          O[(long)(m0 + moff + r) * HD + n0 + n] = (__bf16)acc[mt][nt][r];
      }
    }
  }
}

// ---------------------------------------------------------------------------
// Flash-style causal attention. Block = 8 waves = one 16-row q tile.
__global__ __launch_bounds__(256) void SA_attn_kernel(
    const __bf16* __restrict__ Qb, const __bf16* __restrict__ Kb,
    const __bf16* __restrict__ Vt, float* __restrict__ out) {
  __shared__ __bf16 Qs[16 * HD];        // 32 KB
  __shared__ float  Ss[16 * KC];        //  8 KB
  __shared__ __bf16 Ps[16 * KC];        //  4 KB
  __shared__ float  mrow[16], lrow[16], cfrow[16];

  int blk  = blockIdx.x;
  int b    = blk >> 7;
  int qt   = blk & 127;
  int q0   = qt << 4;
  int tid  = threadIdx.x;
  int wave = tid >> 5;
  int lane = tid & 31;
  int n    = lane & 15;
  int moff = (lane >> 4) << 3;

  // ---- stage Q tile into LDS with one TDM descriptor (16 x 1024 bf16) ----
  if (wave == 0) {
    tdm_load_2d_b16(Qs, Qb + ((long)b * S_LEN + q0) * HD,
                    /*rows=*/16, /*cols=*/HD, /*stride=*/HD);
    __builtin_amdgcn_s_wait_tensorcnt(0);
  }
  if (tid < 16) { mrow[tid] = -3.0e38f; lrow[tid] = 0.f; }
  __syncthreads();

  f32x8 acc[8];
#pragma unroll
  for (int t = 0; t < 8; ++t) acc[t] = (f32x8){0.f,0.f,0.f,0.f,0.f,0.f,0.f,0.f};

  int nchunk = q0 / KC + 1;
  for (int c = 0; c < nchunk; ++c) {
    int kbase = c * KC;
    int col0  = kbase + (wave << 4);

    // ---- stage A: 16x16 score tile per wave (reduce over H) ----
    f32x8 sc = {0.f,0.f,0.f,0.f,0.f,0.f,0.f,0.f};
    if (col0 <= q0 + 15) {              // wave-uniform branch
      const __bf16* Kp = Kb + ((long)b * S_LEN + col0) * HD;
      for (int kk = 0; kk < HD; kk += 32) {
        __builtin_prefetch(Kp + kk + 256, 0, 1);
        bf16x16 a  = load_frag_a(Qs + kk, HD);
        bf16x16 bm = load_frag_b(Kp + kk, HD);
        sc = wmma_bf16(a, bm, sc);
      }
    }
    {
      int col = col0 + n;
#pragma unroll
      for (int r = 0; r < 8; ++r) {
        int qg = q0 + moff + r;
        float v = (col <= qg) ? sc[r] * SCALE : -3.0e38f;
        Ss[(moff + r) * KC + (wave << 4) + n] = v;
      }
    }
    __syncthreads();

    // ---- stage A2: online softmax over this 128-wide chunk ----
    if (tid < 16) {
      int row = tid;
      float mold = mrow[row];
      float mx = mold;
      for (int j = 0; j < KC; ++j) mx = fmaxf(mx, Ss[row * KC + j]);
      float cfv = __expf(mold - mx);
      float sum = 0.f;
      for (int j = 0; j < KC; ++j) {
        float e = __expf(Ss[row * KC + j] - mx);
        sum += e;
        Ps[row * KC + j] = (__bf16)e;
      }
      lrow[row] = lrow[row] * cfv + sum;
      mrow[row] = mx;
      cfrow[row] = cfv;
    }
    __syncthreads();

    // ---- stage B: O[:, wave*128 .. +127] += P(16xKC) x Vt-slice ----
    {
      float cfl[8];
#pragma unroll
      for (int r = 0; r < 8; ++r) cfl[r] = cfrow[moff + r];
#pragma unroll
      for (int t = 0; t < 8; ++t)
#pragma unroll
        for (int r = 0; r < 8; ++r) acc[t][r] *= cfl[r];

      int hbase = wave << 7;
      const __bf16* Vbase = Vt + (long)b * HD * S_LEN + kbase;
      for (int kk = 0; kk < KC; kk += 32) {
        bf16x16 a = load_frag_a(Ps + kk, KC);
#pragma unroll
        for (int t = 0; t < 8; ++t) {
          bf16x16 bm = load_frag_b(Vbase + (long)(hbase + (t << 4)) * S_LEN + kk,
                                   S_LEN);
          acc[t] = wmma_bf16(a, bm, acc[t]);
        }
      }
    }
    __syncthreads();
  }

  // ---- epilogue ----
  float inv_l[8];
#pragma unroll
  for (int r = 0; r < 8; ++r) inv_l[r] = 1.0f / lrow[moff + r];
  int hbase = wave << 7;
#pragma unroll
  for (int t = 0; t < 8; ++t)
#pragma unroll
    for (int r = 0; r < 8; ++r) {
      int qg = q0 + moff + r;
      out[(long)b * S_LEN * HD + (long)qg * HD + hbase + (t << 4) + n] =
          acc[t][r] * inv_l[r];
    }
}

// ---------------------------------------------------------------------------
extern "C" void kernel_launch(void* const* d_in, const int* in_sizes, int n_in,
                              void* d_out, int out_size, void* d_ws, size_t ws_size,
                              hipStream_t stream) {
  const float* X  = (const float*)d_in[0];
  // d_in[1] = attention_mask (causal; recomputed analytically, not read)
  const float* wq = (const float*)d_in[2];
  const float* wk = (const float*)d_in[3];
  const float* wv = (const float*)d_in[4];
  float* out = (float*)d_out;

  char* ws = (char*)d_ws;
  const size_t MB = 1024ull * 1024ull;
  __bf16* Xb  = (__bf16*)(ws);
  __bf16* Wqb = (__bf16*)(ws + 16 * MB);
  __bf16* Wkb = (__bf16*)(ws + 18 * MB);
  __bf16* Wvb = (__bf16*)(ws + 20 * MB);
  __bf16* Qb  = (__bf16*)(ws + 22 * MB);
  __bf16* Kb  = (__bf16*)(ws + 38 * MB);
  __bf16* Vtb = (__bf16*)(ws + 54 * MB);

  int nx4 = (BATCH * S_LEN * HD) / 4;
  int nw4 = (HD * HD) / 4;
  SA_cvt_kernel<<<(nx4 + 255) / 256, 256, 0, stream>>>(X,  Xb,  nx4);
  SA_cvt_kernel<<<(nw4 + 255) / 256, 256, 0, stream>>>(wq, Wqb, nw4);
  SA_cvt_kernel<<<(nw4 + 255) / 256, 256, 0, stream>>>(wk, Wkb, nw4);
  SA_cvt_kernel<<<(nw4 + 255) / 256, 256, 0, stream>>>(wv, Wvb, nw4);

  SA_qkv_kernel<<<3 * 512, 256, 0, stream>>>(Xb, Wqb, Wkb, Wvb, Qb, Kb, Vtb);

  SA_attn_kernel<<<BATCH * (S_LEN / 16), 256, 0, stream>>>(Qb, Kb, Vtb, out);
}
